// S4Classifier_66073776882432
// MI455X (gfx1250) — matrix operand order repaired and checked by reading, not compiled, not run on hardware
//
#include <hip/hip_runtime.h>
#include <hip/hip_bf16.h>
#include <stdint.h>

#define B_  32
#define L_  8192
#define DIN 16
#define H_  128
#define NL_ 4
#define N2_ 32

typedef __attribute__((ext_vector_type(16))) __bf16 v16bf;
typedef __attribute__((ext_vector_type(8)))  float  v8f;
typedef __attribute__((ext_vector_type(4)))  int    v4i;
typedef __attribute__((ext_vector_type(8)))  int    v8i;

// ---------------------------------------------------------------------------
// K1: encoder  h[b,c,l] = x[b,l,:16] @ enc_w[:,c] + enc_b[c]   (h stored B,H,L)
// enc_w/enc_b staged to LDS with GLOBAL_LOAD_ASYNC_TO_LDS_B128 (ASYNCcnt path).
// ---------------------------------------------------------------------------
__global__ void k_encoder(const float* __restrict__ x, const float* __restrict__ enc_w,
                          const float* __restrict__ enc_b, float* __restrict__ h) {
  __shared__ __align__(16) float smem[DIN * H_ + H_];   // ws[2048] ++ bias[128]
  const int tid = threadIdx.x;
  {
    const unsigned lds_base = (unsigned)(uintptr_t)(&smem[0]);
    // enc_w: 8192B = 512 x 16B chunks; thread t copies chunks t and t+256
    unsigned long long ga0 = (unsigned long long)(uintptr_t)(enc_w) + (unsigned)tid * 16u;
    unsigned la0 = lds_base + (unsigned)tid * 16u;
    asm volatile("global_load_async_to_lds_b128 %0, %1, off" :: "v"(la0), "v"(ga0) : "memory");
    unsigned long long ga1 = ga0 + 4096u;
    unsigned la1 = la0 + 4096u;
    asm volatile("global_load_async_to_lds_b128 %0, %1, off" :: "v"(la1), "v"(ga1) : "memory");
    if (tid < 32) {  // enc_b: 512B = 32 x 16B chunks
      unsigned long long ga2 = (unsigned long long)(uintptr_t)(enc_b) + (unsigned)tid * 16u;
      unsigned la2 = lds_base + 8192u + (unsigned)tid * 16u;
      asm volatile("global_load_async_to_lds_b128 %0, %1, off" :: "v"(la2), "v"(ga2) : "memory");
    }
    asm volatile("s_wait_asynccnt 0x0" ::: "memory");
  }
  __syncthreads();
  const float* ws = smem;
  const float* bs = smem + DIN * H_;

  const size_t gid = (size_t)blockIdx.x * 256 + tid;
  const int b = (int)(gid / L_);
  const int l = (int)(gid % L_);
  float xv[DIN];
  const float4* xp = (const float4*)(x + gid * DIN);
#pragma unroll
  for (int q = 0; q < 4; ++q) {
    float4 v = xp[q];
    xv[4 * q + 0] = v.x; xv[4 * q + 1] = v.y; xv[4 * q + 2] = v.z; xv[4 * q + 3] = v.w;
  }
  float* hp = h + (size_t)b * H_ * L_ + l;
  for (int c = 0; c < H_; ++c) {
    float acc = bs[c];
#pragma unroll
    for (int d = 0; d < DIN; ++d) acc = fmaf(xv[d], ws[d * H_ + c], acc);
    hp[(size_t)c * L_] = acc;
  }
}

// ---------------------------------------------------------------------------
// K2: per-layer S4D parameters: lambda = exp(dt*A), Ct = C*(lambda-1)/A
// ---------------------------------------------------------------------------
__global__ void k_params(const float* __restrict__ log_dt, const float* __restrict__ C_re,
                         const float* __restrict__ C_im, const float* __restrict__ log_A_real,
                         const float* __restrict__ A_imag,
                         float* __restrict__ ct_re, float* __restrict__ ct_im,
                         float* __restrict__ lam_re, float* __restrict__ lam_im, int layer) {
  int idx = blockIdx.x * 256 + threadIdx.x;
  if (idx >= H_ * N2_) return;
  int hh = idx / N2_;
  float dt  = expf(log_dt[layer * H_ + hh]);
  int base  = layer * H_ * N2_ + idx;
  float Are = -expf(log_A_real[base]);
  float Aim = A_imag[base];
  float dre = Are * dt, dim = Aim * dt;
  float er  = expf(dre);
  float lr  = er * cosf(dim);
  float li  = er * sinf(dim);
  float nr  = lr - 1.0f, ni = li;
  float inv = 1.0f / (Are * Are + Aim * Aim);
  float qr  = (nr * Are + ni * Aim) * inv;   // (lam-1)/A = (lam-1)*conj(A)/|A|^2
  float qi  = (ni * Are - nr * Aim) * inv;
  float cr  = C_re[base], ci = C_im[base];
  ct_re[idx]  = cr * qr - ci * qi;
  ct_im[idx]  = cr * qi + ci * qr;
  lam_re[idx] = lr;
  lam_im[idx] = li;
}

// ---------------------------------------------------------------------------
// K3: layernorm over H at each (b,l):  u = (h-mu)*rsqrt(var+eps)*g + b
// ---------------------------------------------------------------------------
__global__ void k_layernorm(const float* __restrict__ h, const float* __restrict__ g,
                            const float* __restrict__ be, float* __restrict__ u, int layer) {
  const size_t gid = (size_t)blockIdx.x * 256 + threadIdx.x;
  const int b = (int)(gid / L_);
  const int l = (int)(gid % L_);
  const float* hp = h + (size_t)b * H_ * L_ + l;
  float s = 0.0f, s2 = 0.0f;
  for (int c = 0; c < H_; ++c) {
    float v = hp[(size_t)c * L_];
    s += v; s2 = fmaf(v, v, s2);
  }
  float mu   = s * (1.0f / H_);
  float var  = s2 * (1.0f / H_) - mu * mu;
  float rstd = rsqrtf(var + 1e-5f);
  const float* gg = g + layer * H_;
  const float* bb = be + layer * H_;
  float* up = u + (size_t)b * H_ * L_ + l;
  for (int c = 0; c < H_; ++c) {
    float v = hp[(size_t)c * L_];
    up[(size_t)c * L_] = (v - mu) * rstd * gg[c] + bb[c];
  }
}

// ---------------------------------------------------------------------------
// K4: diagonal-SSM scan (== FFT conv of reference), fused +D*u and exact gelu.
// One wave per (b,c); lane n holds state S_n; butterfly reduce over 32 modes.
// ---------------------------------------------------------------------------
__global__ void k_scan(const float* __restrict__ u, const float* __restrict__ ct_re,
                       const float* __restrict__ ct_im, const float* __restrict__ lam_re,
                       const float* __restrict__ lam_im, const float* __restrict__ Dp,
                       __bf16* __restrict__ yg, int layer) {
  const int wid  = blockIdx.x * 8 + (threadIdx.x >> 5);
  const int lane = threadIdx.x & 31;
  const int b = wid / H_, c = wid % H_;
  const float cr = ct_re[c * N2_ + lane], ci = ct_im[c * N2_ + lane];
  const float lr = lam_re[c * N2_ + lane], li = lam_im[c * N2_ + lane];
  const float dp = Dp[layer * H_ + c];
  const float* up = u + ((size_t)b * H_ + c) * L_;
  __bf16* yp = yg + ((size_t)b * H_ + c) * L_;
  float Sre = 0.0f, Sim = 0.0f;
  for (int l0 = 0; l0 < L_; l0 += 32) {
    float uc = up[l0 + lane];
    __builtin_prefetch(up + l0 + 64, 0, 0);   // global_prefetch_b8
    float mine = 0.0f;
#pragma unroll 8
    for (int t = 0; t < 32; ++t) {
      float uv = __shfl(uc, t, 32);
      float nr = fmaf(lr, Sre, fmaf(-li, Sim, uv));
      float ni = fmaf(lr, Sim, li * Sre);
      Sre = nr; Sim = ni;
      float part = 2.0f * (cr * Sre - ci * Sim);
      part += __shfl_xor(part, 16, 32);
      part += __shfl_xor(part, 8, 32);
      part += __shfl_xor(part, 4, 32);
      part += __shfl_xor(part, 2, 32);
      part += __shfl_xor(part, 1, 32);
      float y   = fmaf(dp, uv, part);
      float gel = 0.5f * y * (1.0f + erff(y * 0.70710678f));
      if (lane == t) mine = gel;
    }
    yp[l0 + lane] = (__bf16)mine;
  }
}

// ---------------------------------------------------------------------------
// K5: fused (out_w @ gelu(y)) + bias -> GLU -> residual-add into h.
// B strip (128 rows x 16 cols bf16, row stride L) pulled into LDS by the
// Tensor Data Mover: one tensor_load_to_lds issued by wave 0 with a 2D D#.
// 8 waves cover all 256 output rows; K=128 in 4 WMMA steps.
// ---------------------------------------------------------------------------
__global__ void k_gemm_glu(const float* __restrict__ W, const float* __restrict__ bias,
                           const __bf16* __restrict__ yg, float* __restrict__ h) {
  const int n0  = blockIdx.x * 16;
  const int b   = blockIdx.y;
  const int tid = threadIdx.x;
  __shared__ __align__(16) __bf16 Bs[H_ * 16];   // [k][n], 4KB

  if (tid < 32) {   // wave 0 issues the TDM descriptor (EXEC ignored by TDM)
    unsigned long long ga = (unsigned long long)(uintptr_t)(yg + ((size_t)b * H_) * L_ + n0);
    unsigned lds_addr = (unsigned)(uintptr_t)(&Bs[0]);
    v4i g0;
    g0.x = 1;                                              // count=1 (valid user D#)
    g0.y = (int)lds_addr;                                  // lds_addr
    g0.z = (int)(unsigned)(ga & 0xFFFFFFFFull);            // global_addr[31:0]
    g0.w = (int)((unsigned)((ga >> 32) & 0x01FFFFFFull)    // global_addr[56:32]
                 | 0x80000000u);                           // type=2 ("image")
    v8i g1;
    g1.s0 = (1 << 16);       // workgroup_mask=0, data_size=1 (2 bytes)
    g1.s1 = (16 << 16);      // atomic_barrier=0 | tensor_dim0.lo16 = 16
    g1.s2 = (128 << 16);     // tensor_dim0.hi16=0 | tensor_dim1.lo16 = 128
    g1.s3 = (16 << 16);      // tensor_dim1.hi16=0 | tile_dim0 = 16
    g1.s4 = 128;             // tile_dim1 = 128 | tile_dim2 = 0
    g1.s5 = L_;              // tensor_dim0_stride.lo32 = 8192 (elements)
    g1.s6 = 0;               // dim0_stride.hi16 | dim1_stride.lo16
    g1.s7 = 0;               // dim1_stride.hi32
    asm volatile("tensor_load_to_lds %0, %1" :: "s"(g0), "s"(g1) : "memory");
    __builtin_amdgcn_s_wait_tensorcnt(0);
  }
  __syncthreads();

  const int wv   = tid >> 5;        // 0..7 -> M tile
  const int lane = tid & 31;
  const int mlo  = lane & 15;
  const int hf   = lane >> 4;
  v8f acc_a = {};
  v8f acc_g = {};
  const float* Wa = W + (size_t)(16 * wv + mlo) * H_;   // a-half rows
  const float* Wg = Wa + (size_t)H_ * H_;               // g-half rows (+128)
#pragma unroll
  for (int k0 = 0; k0 < H_; k0 += 32) {
    v16bf afrag, gfrag, bfrag;
#pragma unroll
    for (int j = 0; j < 8; ++j) {
      // A-layout: VGPR j<4 -> K=2j(+8*hf), j>=4 -> K=16+2(j-4)(+8*hf)
      int kk = (j < 4) ? (2 * j + 8 * hf) : (16 + 2 * (j - 4) + 8 * hf);
      float2 wa = *(const float2*)(Wa + k0 + kk);
      float2 wg = *(const float2*)(Wg + k0 + kk);
      afrag[2 * j]     = (__bf16)wa.x;
      afrag[2 * j + 1] = (__bf16)wa.y;
      gfrag[2 * j]     = (__bf16)wg.x;
      gfrag[2 * j + 1] = (__bf16)wg.y;
      // B-layout: N=lane%16; same K mapping as A (VGPR j: kk, kk+1)
      int kb = k0 + kk;
      bfrag[2 * j]     = Bs[kb * 16 + mlo];
      bfrag[2 * j + 1] = Bs[(kb + 1) * 16 + mlo];
    }
    acc_a = __builtin_amdgcn_wmma_f32_16x16x32_bf16(false, afrag, false, bfrag,
                                                    (short)0, acc_a, false, false);
    acc_g = __builtin_amdgcn_wmma_f32_16x16x32_bf16(false, gfrag, false, bfrag,
                                                    (short)0, acc_g, false, false);
  }
  // C/D layout: VGPR r -> row r + 8*(lane>=16), col = lane%16
#pragma unroll
  for (int r = 0; r < 8; ++r) {
    int m = 16 * wv + r + 8 * hf;
    int n = n0 + mlo;
    float a = acc_a[r] + bias[m];
    float g = acc_g[r] + bias[H_ + m];
    float z = a * (1.0f / (1.0f + expf(-g)));
    h[((size_t)b * H_ + m) * L_ + n] += z;   // residual
  }
}

// ---------------------------------------------------------------------------
// K6: masked sum over L per (b,c)  (one wave per channel)
// ---------------------------------------------------------------------------
__global__ void k_pool(const float* __restrict__ h, const float* __restrict__ mask,
                       float* __restrict__ pooled) {
  const int b    = blockIdx.y;
  const int c    = blockIdx.x * 8 + (threadIdx.x >> 5);
  const int lane = threadIdx.x & 31;
  const float* hp = h + ((size_t)b * H_ + c) * L_;
  const float* mp = mask + (size_t)b * L_;
  float s = 0.0f;
  for (int l = lane; l < L_; l += 32) s = fmaf(hp[l], mp[l], s);
  s += __shfl_xor(s, 16, 32);
  s += __shfl_xor(s, 8, 32);
  s += __shfl_xor(s, 4, 32);
  s += __shfl_xor(s, 2, 32);
  s += __shfl_xor(s, 1, 32);
  if (lane == 0) pooled[b * H_ + c] = s;
}

// ---------------------------------------------------------------------------
// K7: decode: out[b,j] = (pooled[b,:]/(masksum+1e-8)) @ dec_w[:,j] + dec_b[j]
// ---------------------------------------------------------------------------
__global__ void k_decode(const float* __restrict__ pooled, const float* __restrict__ mask,
                         const float* __restrict__ dec_w, const float* __restrict__ dec_b,
                         float* __restrict__ out) {
  int t = threadIdx.x;
  if (t >= B_ * 2) return;
  int b = t >> 1, j = t & 1;
  const float* mp = mask + (size_t)b * L_;
  float ms = 0.0f;
  for (int l = 0; l < L_; ++l) ms += mp[l];
  float inv = 1.0f / (ms + 1e-8f);
  float acc = dec_b[j];
  for (int c = 0; c < H_; ++c) acc = fmaf(pooled[b * H_ + c] * inv, dec_w[c * 2 + j], acc);
  out[t] = acc;
}

// ---------------------------------------------------------------------------
extern "C" void kernel_launch(void* const* d_in, const int* in_sizes, int n_in,
                              void* d_out, int out_size, void* d_ws, size_t ws_size,
                              hipStream_t stream) {
  (void)in_sizes; (void)n_in; (void)out_size; (void)ws_size;
  const float* x      = (const float*)d_in[0];
  const float* mask   = (const float*)d_in[1];
  const float* enc_w  = (const float*)d_in[2];
  const float* enc_b  = (const float*)d_in[3];
  const float* ln_g   = (const float*)d_in[4];
  const float* ln_b   = (const float*)d_in[5];
  const float* log_dt = (const float*)d_in[6];
  const float* C_re   = (const float*)d_in[7];
  const float* C_im   = (const float*)d_in[8];
  const float* logA   = (const float*)d_in[9];
  const float* A_im   = (const float*)d_in[10];
  const float* Dp     = (const float*)d_in[11];
  const float* out_w  = (const float*)d_in[12];
  const float* out_b  = (const float*)d_in[13];
  const float* dec_w  = (const float*)d_in[14];
  const float* dec_b  = (const float*)d_in[15];

  char* ws = (char*)d_ws;
  const size_t SZ_HL = (size_t)B_ * H_ * L_;              // 33.5M elems
  float*  h      = (float*)(ws);                          // 128 MB
  float*  u      = (float*)(ws + SZ_HL * 4);              // 128 MB
  __bf16* yg     = (__bf16*)(ws + SZ_HL * 8);             //  64 MB
  float*  ct_re  = (float*)(ws + SZ_HL * 10);
  float*  ct_im  = ct_re  + H_ * N2_;
  float*  lam_re = ct_im  + H_ * N2_;
  float*  lam_im = lam_re + H_ * N2_;
  float*  pooled = lam_im + H_ * N2_;

  k_encoder<<<dim3((unsigned)((size_t)B_ * L_ / 256)), 256, 0, stream>>>(x, enc_w, enc_b, h);

  for (int i = 0; i < NL_; ++i) {
    k_params<<<dim3((H_ * N2_ + 255) / 256), 256, 0, stream>>>(
        log_dt, C_re, C_im, logA, A_im, ct_re, ct_im, lam_re, lam_im, i);
    k_layernorm<<<dim3((unsigned)((size_t)B_ * L_ / 256)), 256, 0, stream>>>(h, ln_g, ln_b, u, i);
    k_scan<<<dim3(B_ * H_ / 8), 256, 0, stream>>>(u, ct_re, ct_im, lam_re, lam_im, Dp, yg, i);
    k_gemm_glu<<<dim3(L_ / 16, B_), 256, 0, stream>>>(
        out_w + (size_t)i * 2 * H_ * H_, out_b + (size_t)i * 2 * H_, yg, h);
  }

  k_pool<<<dim3(H_ / 8, B_), 256, 0, stream>>>(h, mask, pooled);
  k_decode<<<1, 64, 0, stream>>>(pooled, mask, dec_w, dec_b, (float*)d_out);
}
